// GN_72559177498912
// MI455X (gfx1250) — matrix operand order repaired and checked by reading, not compiled, not run on hardware
//
#include <hip/hip_runtime.h>
#include <hip/hip_bf16.h>

typedef __bf16 bf16_t;
typedef __attribute__((ext_vector_type(16))) __bf16 v16bf;
typedef __attribute__((ext_vector_type(8)))  __bf16 v8bf;
typedef __attribute__((ext_vector_type(4)))  __bf16 v4bf;
typedef __attribute__((ext_vector_type(8)))  float  v8f;
typedef __attribute__((ext_vector_type(4)))  float  v4f;

#define GN_N_NODES 50000
#define GN_N_EDGES 800000
#define GN_NF 128
#define GN_EF 64
#define GN_H  256
#define GN_EO 128
#define GN_NO 128

union FragA { v16bf v; v8bf h[2]; };

// One 16x64 output strip (4 n-tiles) per wave: A fragment loaded once per k-step,
// reused by 4 v_wmma_f32_16x16x32_bf16 into 4 accumulators.
// Xlds: row-major [M][ldx] bf16 activations in LDS.
// WT:   row-major [N][K] bf16 (weights transposed) in global (L2/WGP$-resident).
// A order used here (contiguous k): lanes 0-15 row=lane hold K {0..15}; lanes 16-31 hold K {16..31}.
// B order used here (same k order as A): lane%16 = column; lanes 0-15 elements j -> K=j, lanes 16-31 -> K=16+j.
__device__ __forceinline__ void gn_gemm_strip4(const bf16_t* Xlds, int ldx, int mbase,
                                               const bf16_t* __restrict__ WT, int K,
                                               int nbase, int lane, v8f acc[4]) {
  const int hl = lane & 15;
  const bf16_t* xrow = Xlds + (mbase + hl) * ldx + ((lane < 16) ? 0 : 16);
  const bf16_t* wrow = WT + (size_t)(nbase + hl) * K + ((lane < 16) ? 0 : 16);
  v8f z = {};
  acc[0] = z; acc[1] = z; acc[2] = z; acc[3] = z;
  #pragma unroll 2
  for (int k0 = 0; k0 < K; k0 += 32) {
    FragA a;
    a.h[0] = *(const v8bf*)(xrow + k0);        // K = k0 + koff + 0..7
    a.h[1] = *(const v8bf*)(xrow + k0 + 8);    // K = k0 + koff + 8..15
    #pragma unroll
    for (int n = 0; n < 4; ++n) {
      v16bf bv = *(const v16bf*)(wrow + (size_t)(n * 16) * K + k0);  // 32B contiguous
      acc[n] = __builtin_amdgcn_wmma_f32_16x16x32_bf16(false, a.v, false, bv,
                                                       (short)0, acc[n], false, false);
    }
  }
}

// C/D layout: lanes 0-15 col=tile_base+lane rows mbase+0..7 in c[0..7];
// lanes 16-31 col=tile_base+lane-16 rows mbase+8..15.  Bias + ReLU -> bf16 LDS.
__device__ __forceinline__ void gn_store_act_strip4(bf16_t* Hlds, int ldh, int mbase, int nbase,
                                                    int lane, const v8f* acc,
                                                    const float* __restrict__ bias) {
  const int r0 = mbase + ((lane < 16) ? 0 : 8);
  #pragma unroll
  for (int n = 0; n < 4; ++n) {
    const int col = nbase + n * 16 + (lane & 15);
    const float b = bias[col];
    #pragma unroll
    for (int r = 0; r < 8; ++r) {
      float v = acc[n][r] + b;
      Hlds[(r0 + r) * ldh + col] = (bf16_t)(v > 0.0f ? v : 0.0f);
    }
  }
}

// ---------------- Edge MLP: gather -> 320->256->256->128, fused scatter-add ----------------
__global__ __launch_bounds__(256) void gn_edge_mlp(
    const float* __restrict__ node_attrs, const float* __restrict__ edge_attrs,
    const int* __restrict__ senders, const int* __restrict__ receivers,
    const bf16_t* __restrict__ W0T, const float* __restrict__ b0,
    const bf16_t* __restrict__ W1T, const float* __restrict__ b1,
    const bf16_t* __restrict__ W2T, const float* __restrict__ b2,
    float* __restrict__ e_out, float* __restrict__ acc)
{
  __shared__ __align__(32) bf16_t smem[64 * 320 + 64 * 256];
  bf16_t* X  = smem;              // [64][320]  layer-0 input
  bf16_t* Hp = smem + 64 * 320;   // [64][256]  ping
  bf16_t* Hq = smem;              // [64][256]  pong (X dead after layer 0)

  const int tid  = threadIdx.x;
  const int lane = tid & 31;
  const int wave = tid >> 5;
  const int ebase = blockIdx.x * 64;

  // Warm L2/WGP$ with layer-0 weights (256 rows x 320 bf16 = 160 KB, shared by all blocks)
  __builtin_prefetch(W0T + (size_t)tid * 320, 0, 3);

  // Gather: 64 edges x 80 float4 (sender 32 | receiver 32 | edge 16), cvt to bf16 in LDS
  for (int i = tid; i < 64 * 80; i += 256) {
    const int e = i / 80;
    const int q = i - e * 80;
    const int ge = ebase + e;
    v4f v;
    if (q < 32)       v = ((const v4f*)(node_attrs + (size_t)senders[ge]   * GN_NF))[q];
    else if (q < 64)  v = ((const v4f*)(node_attrs + (size_t)receivers[ge] * GN_NF))[q - 32];
    else              v = ((const v4f*)(edge_attrs + (size_t)ge * GN_EF))[q - 64];
    v4bf o;
    o[0] = (bf16_t)v.x; o[1] = (bf16_t)v.y; o[2] = (bf16_t)v.z; o[3] = (bf16_t)v.w;
    *(v4bf*)(X + e * 320 + q * 4) = o;
  }
  __syncthreads();

  // Layer 0: [64x320] @ [320x256] + ReLU   (16 strips of 16x64 over 8 waves)
  for (int t = wave; t < 16; t += 8) {
    const int mt = (t & 3) << 4, ns = (t >> 2) << 6;
    v8f c[4];
    gn_gemm_strip4(X, 320, mt, W0T, 320, ns, lane, c);
    gn_store_act_strip4(Hp, 256, mt, ns, lane, c, b0);
  }
  __syncthreads();

  // Layer 1: [64x256] @ [256x256] + ReLU
  for (int t = wave; t < 16; t += 8) {
    const int mt = (t & 3) << 4, ns = (t >> 2) << 6;
    v8f c[4];
    gn_gemm_strip4(Hp, 256, mt, W1T, 256, ns, lane, c);
    gn_store_act_strip4(Hq, 256, mt, ns, lane, c, b1);
  }
  __syncthreads();

  // Layer 2: [64x256] @ [256x128]; write e_out + fused segment-sum (f32 atomics in L2)
  {
    const int mt = (wave & 3) << 4, ns = (wave >> 2) << 6;   // 8 strips, one per wave
    v8f c[4];
    gn_gemm_strip4(Hq, 256, mt, W2T, 256, ns, lane, c);
    const int r0 = mt + ((lane < 16) ? 0 : 8);
    #pragma unroll
    for (int n = 0; n < 4; ++n) {
      const int col = ns + n * 16 + (lane & 15);
      const float bb = b2[col];
      #pragma unroll
      for (int r = 0; r < 8; ++r) {
        const int ge = ebase + r0 + r;
        const float v = c[n][r] + bb;
        e_out[(size_t)ge * GN_EO + col] = v;
        __hip_atomic_fetch_add(acc + (size_t)receivers[ge] * GN_EO + col, v,
                               __ATOMIC_RELAXED, __HIP_MEMORY_SCOPE_AGENT);
      }
    }
  }
}

// ---------------- Node MLP: concat(node_attrs, agg) -> 256->256->256->128 ----------------
__global__ __launch_bounds__(256) void gn_node_mlp(
    const float* __restrict__ node_attrs, const float* __restrict__ acc,
    const bf16_t* __restrict__ W0T, const float* __restrict__ b0,
    const bf16_t* __restrict__ W1T, const float* __restrict__ b1,
    const bf16_t* __restrict__ W2T, const float* __restrict__ b2,
    float* __restrict__ n_out)
{
  __shared__ __align__(32) bf16_t smem[64 * 256 + 64 * 256];
  bf16_t* X  = smem;              // [64][256]
  bf16_t* Hp = smem + 64 * 256;   // ping
  bf16_t* Hq = smem;              // pong (X dead after layer 0)

  const int tid  = threadIdx.x;
  const int lane = tid & 31;
  const int wave = tid >> 5;
  const int nb = blockIdx.x * 64;

  __builtin_prefetch(W0T + (size_t)tid * 256, 0, 3);

  // Gather: 64 rows x 64 float4 (node_attrs 32 | agg 32); zero-pad OOB tail rows
  for (int i = tid; i < 64 * 64; i += 256) {
    const int r = i >> 6;
    const int q = i & 63;
    const int gr = nb + r;
    v4f v = {0.f, 0.f, 0.f, 0.f};
    if (gr < GN_N_NODES) {
      if (q < 32) v = ((const v4f*)(node_attrs + (size_t)gr * GN_NF))[q];
      else        v = ((const v4f*)(acc        + (size_t)gr * GN_EO))[q - 32];
    }
    v4bf o;
    o[0] = (bf16_t)v.x; o[1] = (bf16_t)v.y; o[2] = (bf16_t)v.z; o[3] = (bf16_t)v.w;
    *(v4bf*)(X + r * 256 + q * 4) = o;
  }
  __syncthreads();

  for (int t = wave; t < 16; t += 8) {
    const int mt = (t & 3) << 4, ns = (t >> 2) << 6;
    v8f c[4];
    gn_gemm_strip4(X, 256, mt, W0T, 256, ns, lane, c);
    gn_store_act_strip4(Hp, 256, mt, ns, lane, c, b0);
  }
  __syncthreads();
  for (int t = wave; t < 16; t += 8) {
    const int mt = (t & 3) << 4, ns = (t >> 2) << 6;
    v8f c[4];
    gn_gemm_strip4(Hp, 256, mt, W1T, 256, ns, lane, c);
    gn_store_act_strip4(Hq, 256, mt, ns, lane, c, b1);
  }
  __syncthreads();
  {
    const int mt = (wave & 3) << 4, ns = (wave >> 2) << 6;   // 64x128 out: one strip per wave
    v8f c[4];
    gn_gemm_strip4(Hq, 256, mt, W2T, 256, ns, lane, c);
    const int r0 = mt + ((lane < 16) ? 0 : 8);
    #pragma unroll
    for (int n = 0; n < 4; ++n) {
      const int col = ns + n * 16 + (lane & 15);
      const float bb = b2[col];
      #pragma unroll
      for (int r = 0; r < 8; ++r) {
        const int gr = nb + r0 + r;
        if (gr < GN_N_NODES) n_out[(size_t)gr * GN_NO + col] = c[n][r] + bb;
      }
    }
  }
}

// -------- helpers: weight transpose+bf16 convert, zero aggregation buffer --------
__global__ void gn_w_to_bf16_t(const float* __restrict__ src, bf16_t* __restrict__ dst,
                               int K, int N) {
  const int idx = blockIdx.x * blockDim.x + threadIdx.x;
  if (idx < K * N) {
    const int k = idx / N, n = idx - k * N;
    dst[(size_t)n * K + k] = (bf16_t)src[idx];   // [K][N] f32 -> [N][K] bf16
  }
}

__global__ void gn_zero_f32(float* __restrict__ p, int n4) {
  const int i = blockIdx.x * blockDim.x + threadIdx.x;
  if (i < n4) ((v4f*)p)[i] = (v4f){0.f, 0.f, 0.f, 0.f};
}

extern "C" void kernel_launch(void* const* d_in, const int* in_sizes, int n_in,
                              void* d_out, int out_size, void* d_ws, size_t ws_size,
                              hipStream_t stream) {
  const float* node_attrs = (const float*)d_in[0];
  const float* edge_attrs = (const float*)d_in[1];
  const int*   senders    = (const int*)d_in[2];
  const int*   receivers  = (const int*)d_in[3];
  const float* fe_W0 = (const float*)d_in[4];  const float* fe_b0 = (const float*)d_in[5];
  const float* fe_W1 = (const float*)d_in[6];  const float* fe_b1 = (const float*)d_in[7];
  const float* fe_W2 = (const float*)d_in[8];  const float* fe_b2 = (const float*)d_in[9];
  const float* fn_W0 = (const float*)d_in[10]; const float* fn_b0 = (const float*)d_in[11];
  const float* fn_W1 = (const float*)d_in[12]; const float* fn_b1 = (const float*)d_in[13];
  const float* fn_W2 = (const float*)d_in[14]; const float* fn_b2 = (const float*)d_in[15];

  float* n_out = (float*)d_out;
  float* e_out = n_out + (size_t)GN_N_NODES * GN_NO;

  // workspace: f32 aggregation buffer + 6 transposed bf16 weight matrices
  char* ws = (char*)d_ws;
  size_t off = 0;
  float* acc = (float*)(ws + off); off += (size_t)GN_N_NODES * GN_EO * sizeof(float);
  off = (off + 255) & ~(size_t)255;
  bf16_t* feW0T = (bf16_t*)(ws + off); off += (size_t)256 * 320 * sizeof(bf16_t); off = (off + 255) & ~(size_t)255;
  bf16_t* feW1T = (bf16_t*)(ws + off); off += (size_t)256 * 256 * sizeof(bf16_t); off = (off + 255) & ~(size_t)255;
  bf16_t* feW2T = (bf16_t*)(ws + off); off += (size_t)128 * 256 * sizeof(bf16_t); off = (off + 255) & ~(size_t)255;
  bf16_t* fnW0T = (bf16_t*)(ws + off); off += (size_t)256 * 256 * sizeof(bf16_t); off = (off + 255) & ~(size_t)255;
  bf16_t* fnW1T = (bf16_t*)(ws + off); off += (size_t)256 * 256 * sizeof(bf16_t); off = (off + 255) & ~(size_t)255;
  bf16_t* fnW2T = (bf16_t*)(ws + off); off += (size_t)128 * 256 * sizeof(bf16_t);

  // Stage weights as bf16-transposed (deterministic each call) + zero aggregation buffer
  gn_w_to_bf16_t<<<(320 * 256 + 255) / 256, 256, 0, stream>>>(fe_W0, feW0T, 320, 256);
  gn_w_to_bf16_t<<<(256 * 256 + 255) / 256, 256, 0, stream>>>(fe_W1, feW1T, 256, 256);
  gn_w_to_bf16_t<<<(256 * 128 + 255) / 256, 256, 0, stream>>>(fe_W2, feW2T, 256, 128);
  gn_w_to_bf16_t<<<(256 * 256 + 255) / 256, 256, 0, stream>>>(fn_W0, fnW0T, 256, 256);
  gn_w_to_bf16_t<<<(256 * 256 + 255) / 256, 256, 0, stream>>>(fn_W1, fnW1T, 256, 256);
  gn_w_to_bf16_t<<<(256 * 128 + 255) / 256, 256, 0, stream>>>(fn_W2, fnW2T, 256, 128);
  {
    const int n4 = GN_N_NODES * GN_EO / 4;
    gn_zero_f32<<<(n4 + 255) / 256, 256, 0, stream>>>(acc, n4);
  }

  // Edge MLP with fused segment-sum (800000 / 64 = 12500 blocks, exact)
  gn_edge_mlp<<<GN_N_EDGES / 64, 256, 0, stream>>>(
      node_attrs, edge_attrs, senders, receivers,
      feW0T, fe_b0, feW1T, fe_b1, feW2T, fe_b2, e_out, acc);

  // Node MLP
  gn_node_mlp<<<(GN_N_NODES + 63) / 64, 256, 0, stream>>>(
      node_attrs, acc, fnW0T, fn_b0, fnW1T, fn_b1, fnW2T, fn_b2, n_out);
}
